// GraphTransformer_86517821211097
// MI455X (gfx1250) — matrix operand order repaired and checked by reading, not compiled, not run on hardware
//
#include <hip/hip_runtime.h>
#include <hip/hip_bf16.h>

// ---------------------------------------------------------------------------
// GraphTransformer fused pipeline for gfx1250 (MI455X), bf16 WMMA everywhere.
// Shapes: C=64 G=8 N=32 D=2048 O=256 P=5
// ---------------------------------------------------------------------------

typedef __bf16 bf16;
typedef __attribute__((ext_vector_type(16))) __bf16 v16bf;
typedef __attribute__((ext_vector_type(8)))  float  v8f;

#define NC 64
#define NG 8
#define NN 32
#define DD 2048
#define OO 256
#define PP 5
#define MROWS (NC*NG*NN)   // 16384
#define PPAD  384          // 320 padded to a multiple of 128

__device__ __forceinline__ v8f wmma_bf16(v16bf a, v16bf b, v8f c) {
  return __builtin_amdgcn_wmma_f32_16x16x32_bf16(
      /*neg_a=*/false, a, /*neg_b=*/false, b,
      /*c_mod=*/(short)0, c, /*reuse_a=*/false, /*reuse_b=*/false);
}

// A-matrix 16x32 (MxK), 16-bit: lanes 0-15 row M=lane, elems 0..15 -> K 0..15;
// lanes 16-31 same M, K 16..31.
__device__ __forceinline__ v16bf load_frag_a(const bf16* p0, int lane, int ld) {
  const int m = lane & 15;
  const int kh = (lane >> 4) << 4;
  const bf16* p = p0 + m * ld;
  v16bf f;
#pragma unroll
  for (int e = 0; e < 8; ++e) f[e] = p[kh + e];
#pragma unroll
  for (int e = 0; e < 8; ++e) f[8 + e] = p[kh + 8 + e];
  return f;
}

// B-matrix 32x16 (KxN) sourced from an [N][K] row-major tile (i.e. W rows or a
// pre-transposed X tile): lane gives N = lane&15, lanes>=16 cover K 16..31.
__device__ __forceinline__ v16bf load_frag_b(const bf16* p0, int lane, int ld) {
  const int n = lane & 15;
  const int kb = (lane >> 4) << 4;
  const bf16* p = p0 + n * ld + kb;
  v16bf f;
#pragma unroll
  for (int e = 0; e < 16; ++e) f[e] = p[e];
  return f;
}

// CDNA5 async global->LDS copy (ASYNCcnt-tracked), 16 bytes per lane.
__device__ __forceinline__ void async_copy_b128(void* lds, const void* gptr) {
  const unsigned l = (unsigned)(uintptr_t)lds;   // low 32 bits = LDS offset
  asm volatile("global_load_async_to_lds_b128 %0, %1, off"
               :: "v"(l), "v"(gptr) : "memory");
}
__device__ __forceinline__ void wait_async0() {
  asm volatile("s_wait_asynccnt 0x0" ::: "memory");
}

// ---------------------------------------------------------------------------
// fp32 -> bf16 conversion
// ---------------------------------------------------------------------------
__global__ void k_cvt_bf16(const float* __restrict__ src, bf16* __restrict__ dst, long n) {
  long i = (long)blockIdx.x * blockDim.x + threadIdx.x;
  long stride = (long)gridDim.x * blockDim.x;
  for (; i < n; i += stride) dst[i] = (bf16)src[i];
}

// ---------------------------------------------------------------------------
// Generic GEMM: C[M,N] = epilogue(A[M,K] @ W[N,K]^T)
//   EPI 0: C_bf16 = bf16(acc)
//   EPI 1: C_bf16 = bf16(base_f32 + relu(acc))     (feats = X + relu(ctx@transT))
// 256 thr / 8 waves, 128x128 block tile, wave -> 32x64 (2x4 WMMA tiles).
// K stepped by 32 through double-buffered LDS filled with async-to-LDS copies:
// transfer of stage k+1 overlaps WMMA of stage k. M,N multiples of 128.
// ---------------------------------------------------------------------------
template <int EPI>
__global__ __launch_bounds__(256) void k_gemm_bf16(
    const bf16* __restrict__ A, const bf16* __restrict__ W,
    const float* __restrict__ base, bf16* __restrict__ Cb,
    int M, int N, int K) {
  __shared__ bf16 As[2][128 * 40];
  __shared__ bf16 Ws[2][128 * 40];
  const int tid = threadIdx.x;
  const int lane = tid & 31;
  const int w = tid >> 5;            // 0..7
  const int m0 = blockIdx.y * 128;
  const int n0 = blockIdx.x * 128;
  const int mw = (w >> 1) << 5;      // 0,32,64,96
  const int nw = (w & 1) << 6;       // 0,64

  v8f acc[2][4] = {};

  // stage one 128x32 A tile + 128x32 W tile: 512 x 16B chunks each,
  // 256 threads -> 2 chunks of A and 2 of W per thread, all async.
  auto issue_stage = [&](int buf, int kt) {
#pragma unroll
    for (int t = 0; t < 2; ++t) {
      const int c = tid + t * 256;       // 0..511
      const int r = c >> 2, q = c & 3;   // row, 16B quarter
      async_copy_b128(&As[buf][r * 40 + q * 8], &A[(long)(m0 + r) * K + kt + q * 8]);
      async_copy_b128(&Ws[buf][r * 40 + q * 8], &W[(long)(n0 + r) * K + kt + q * 8]);
    }
  };

  issue_stage(0, 0);
  wait_async0();
  __syncthreads();

  const int nk = K >> 5;
  for (int kt = 0; kt < nk; ++kt) {
    const int cur = kt & 1;
    if (kt + 1 < nk) issue_stage(cur ^ 1, (kt + 1) << 5);

    v16bf af[2], bfr[4];
#pragma unroll
    for (int mi = 0; mi < 2; ++mi)
      af[mi] = load_frag_a(&As[cur][(mw + mi * 16) * 40], lane, 40);
#pragma unroll
    for (int ni = 0; ni < 4; ++ni)
      bfr[ni] = load_frag_b(&Ws[cur][(nw + ni * 16) * 40], lane, 40);
#pragma unroll
    for (int mi = 0; mi < 2; ++mi)
#pragma unroll
      for (int ni = 0; ni < 4; ++ni)
        acc[mi][ni] = wmma_bf16(af[mi], bfr[ni], acc[mi][ni]);

    if (kt + 1 < nk) {
      wait_async0();     // stage kt+1 transfer done (own lanes)
      __syncthreads();   // all waves: reads of 'cur' done + next buffer ready
    }
  }

  const int colb = n0 + nw + (lane & 15);
  const int rowb = m0 + mw + ((lane >> 4) << 3);
#pragma unroll
  for (int mi = 0; mi < 2; ++mi)
#pragma unroll
    for (int ni = 0; ni < 4; ++ni)
#pragma unroll
      for (int v = 0; v < 8; ++v) {
        const long row = rowb + mi * 16 + v;
        const long col = colb + ni * 16;
        float x = acc[mi][ni][v];
        if (EPI == 1) {
          x = base[row * N + col] + (x > 0.f ? x : 0.f);
        }
        Cb[row * N + col] = (bf16)x;
      }
}

// ---------------------------------------------------------------------------
// Inner attention, one block per (c,g):
//   S = (e e^T) / sqrt(O); att = softmax_rows(S); ctx = att @ X
// ---------------------------------------------------------------------------
__global__ __launch_bounds__(128) void k_inner_attn(
    const bf16* __restrict__ Eb, const bf16* __restrict__ Xb,
    bf16* __restrict__ Ctx) {
  __shared__ bf16 eb[32 * 264];
  __shared__ float Sf[32 * 33];
  __shared__ bf16 attb[32 * 40];
  __shared__ bf16 Xt[64 * 40];
  const int tid = threadIdx.x, lane = tid & 31, w = tid >> 5;
  const long cg = blockIdx.x;
  const bf16* E = Eb + cg * 32 * OO;
  const bf16* X = Xb + cg * 32 * DD;

  // stage e [32,256] into LDS
  for (int i = tid; i < 1024; i += 128) {
    const int r = i >> 5, q = i & 31;
    *(int4*)&eb[r * 264 + q * 8] = *(const int4*)&E[r * OO + q * 8];
  }
  __syncthreads();

  // S = e e^T * 1/16 ; wave w owns 16x16 tile (mi=w>>1, ni=w&1)
  {
    const int mi = (w >> 1) << 4, ni = (w & 1) << 4;
    v8f acc = {};
    for (int kt = 0; kt < OO; kt += 32) {
      v16bf a = load_frag_a(&eb[mi * 264 + kt], lane, 264);
      v16bf b = load_frag_b(&eb[ni * 264 + kt], lane, 264);
      acc = wmma_bf16(a, b, acc);
    }
    const int col = ni + (lane & 15);
    const int rb = mi + ((lane >> 4) << 3);
#pragma unroll
    for (int v = 0; v < 8; ++v) Sf[(rb + v) * 33 + col] = acc[v] * 0.0625f;
  }
  __syncthreads();

  // row softmax -> attb (bf16)
  if (tid < 32) {
    float mx = -1e30f;
    for (int m = 0; m < 32; ++m) mx = fmaxf(mx, Sf[tid * 33 + m]);
    float s = 0.f;
    for (int m = 0; m < 32; ++m) s += __expf(Sf[tid * 33 + m] - mx);
    const float inv = 1.f / s;
    for (int m = 0; m < 32; ++m)
      attb[tid * 40 + m] = (bf16)(__expf(Sf[tid * 33 + m] - mx) * inv);
  }
  __syncthreads();

  v16bf af0 = load_frag_a(&attb[0], lane, 40);
  v16bf af1 = load_frag_a(&attb[16 * 40], lane, 40);

  // ctx = att @ X, K=32 single WMMA step, stream d in chunks of 64
  for (int d0 = 0; d0 < DD; d0 += 64) {
    for (int i = tid; i < 2048; i += 128) {
      const int k = i >> 6, n = i & 63;           // coalesced along n (=d)
      Xt[n * 40 + k] = X[k * DD + d0 + n];
    }
    __syncthreads();
    v16bf b = load_frag_b(&Xt[(w << 4) * 40], lane, 40);
    const int col = d0 + (w << 4) + (lane & 15);
    const int rb = (lane >> 4) << 3;
    {
      v8f acc = {};
      acc = wmma_bf16(af0, b, acc);
#pragma unroll
      for (int v = 0; v < 8; ++v)
        Ctx[(cg * 32 + rb + v) * DD + col] = (bf16)acc[v];
    }
    {
      v8f acc = {};
      acc = wmma_bf16(af1, b, acc);
#pragma unroll
      for (int v = 0; v < 8; ++v)
        Ctx[(cg * 32 + 16 + rb + v) * DD + col] = (bf16)acc[v];
    }
    __syncthreads();
  }
}

// ---------------------------------------------------------------------------
// Inter attention, one block per class c:
//   S2 = (p_e @ fa_e^T)/16 ; att2 = softmax_rows(S2) ; out = att2 @ feats[c]
// P=5 padded to 16 rows.
// ---------------------------------------------------------------------------
__global__ __launch_bounds__(128) void k_inter_attn(
    const bf16* __restrict__ PEb, const bf16* __restrict__ FAEb,
    const bf16* __restrict__ Fb, float* __restrict__ Out) {
  __shared__ bf16 peb[16 * 264];
  __shared__ float S2[16 * 260];
  __shared__ bf16 att2b[16 * 264];
  __shared__ bf16 Ft[64 * 40];
  const int tid = threadIdx.x, lane = tid & 31, w = tid >> 5;
  const long c = blockIdx.x;

  for (int i = tid; i < 16 * 264; i += 128) { peb[i] = (bf16)0.f; att2b[i] = (bf16)0.f; }
  __syncthreads();
  const bf16* PE = PEb + c * PP * OO;
  for (int i = tid; i < PP * 32; i += 128) {         // 5 rows x 32 int4
    const int r = i >> 5, q = i & 31;
    *(int4*)&peb[r * 264 + q * 8] = *(const int4*)&PE[r * OO + q * 8];
  }
  __syncthreads();

  // S2: M=16, N=256, K=256; wave w handles n-tiles w, w+4, w+8, w+12
  const bf16* FAE = FAEb + c * OO * OO;
  for (int nt = w; nt < 16; nt += 4) {
    v8f acc = {};
    for (int kt = 0; kt < OO; kt += 32) {
      v16bf a = load_frag_a(&peb[kt], lane, 264);
      v16bf b = load_frag_b(FAE + (long)(nt << 4) * OO + kt, lane, OO);
      acc = wmma_bf16(a, b, acc);
    }
    const int col = (nt << 4) + (lane & 15);
    const int rb = (lane >> 4) << 3;
#pragma unroll
    for (int v = 0; v < 8; ++v) S2[(rb + v) * 260 + col] = acc[v] * 0.0625f;
  }
  __syncthreads();

  if (tid < PP) {
    float mx = -1e30f;
    for (int n = 0; n < OO; ++n) mx = fmaxf(mx, S2[tid * 260 + n]);
    float s = 0.f;
    for (int n = 0; n < OO; ++n) s += __expf(S2[tid * 260 + n] - mx);
    const float inv = 1.f / s;
    for (int n = 0; n < OO; ++n)
      att2b[tid * 264 + n] = (bf16)(__expf(S2[tid * 260 + n] - mx) * inv);
  }
  __syncthreads();

  v16bf af[8];
#pragma unroll
  for (int kt = 0; kt < 8; ++kt) af[kt] = load_frag_a(&att2b[kt * 32], lane, 264);

  const bf16* F = Fb + c * (long)OO * DD;
  for (int d0 = 0; d0 < DD; d0 += 64) {
    v8f acc = {};
    for (int kt = 0; kt < 8; ++kt) {
      for (int i = tid; i < 2048; i += 128) {
        const int k = i >> 6, n = i & 63;
        Ft[n * 40 + k] = F[(long)(kt * 32 + k) * DD + d0 + n];
      }
      __syncthreads();
      v16bf b = load_frag_b(&Ft[(w << 4) * 40], lane, 40);
      acc = wmma_bf16(af[kt], b, acc);
      __syncthreads();
    }
    const int col = d0 + (w << 4) + (lane & 15);
    const int rb = (lane >> 4) << 3;
#pragma unroll
    for (int v = 0; v < 8; ++v) {
      const int row = rb + v;
      if (row < PP) Out[(c * PP + row) * DD + col] = acc[v];
    }
  }
}

// ---------------------------------------------------------------------------
// Host side
// ---------------------------------------------------------------------------
static inline size_t align_up(size_t x, size_t a) { return (x + a - 1) & ~(a - 1); }

extern "C" void kernel_launch(void* const* d_in, const int* in_sizes, int n_in,
                              void* d_out, int out_size, void* d_ws, size_t ws_size,
                              hipStream_t stream) {
  const float* topk  = (const float*)d_in[0];  // [64,8,32,2048]
  const float* proto = (const float*)d_in[1];  // [64,5,2048]
  const float* w1    = (const float*)d_in[2];  // [256,2048]
  const float* trans = (const float*)d_in[3];  // [2048,2048]
  const float* iw1   = (const float*)d_in[4];  // [256,2048]
  const float* iw2   = (const float*)d_in[5];  // [256,2048]
  float* out = (float*)d_out;                  // [64,5,2048]

  char* ws = (char*)d_ws;
  size_t off = 0;
  auto carve = [&](size_t bytes) { char* p = ws + off; off = align_up(off + bytes, 256); return p; };

  bf16* Xb     = (bf16*)carve((size_t)MROWS * DD * 2);   // topk bf16
  bf16* W1b    = (bf16*)carve((size_t)OO * DD * 2);
  bf16* Trb    = (bf16*)carve((size_t)DD * DD * 2);
  bf16* IW1b   = (bf16*)carve((size_t)OO * DD * 2);
  bf16* IW2b   = (bf16*)carve((size_t)OO * DD * 2);
  bf16* Pb     = (bf16*)carve((size_t)PPAD * DD * 2);    // 320 used, 384 padded
  bf16* Eb     = (bf16*)carve((size_t)MROWS * OO * 2);
  bf16* Ctxb   = (bf16*)carve((size_t)MROWS * DD * 2);
  bf16* Featsb = (bf16*)carve((size_t)MROWS * DD * 2);
  bf16* FAEb   = (bf16*)carve((size_t)MROWS * OO * 2);
  bf16* PEb    = (bf16*)carve((size_t)PPAD * OO * 2);    // 320 used, 384 padded
  (void)ws_size;

  // 1) convert everything to bf16
  k_cvt_bf16<<<4096, 256, 0, stream>>>(topk,  Xb,   (long)MROWS * DD);
  k_cvt_bf16<<<512,  256, 0, stream>>>(w1,    W1b,  (long)OO * DD);
  k_cvt_bf16<<<2048, 256, 0, stream>>>(trans, Trb,  (long)DD * DD);
  k_cvt_bf16<<<512,  256, 0, stream>>>(iw1,   IW1b, (long)OO * DD);
  k_cvt_bf16<<<512,  256, 0, stream>>>(iw2,   IW2b, (long)OO * DD);
  k_cvt_bf16<<<512,  256, 0, stream>>>(proto, Pb,   (long)NC * PP * DD);

  // 2) e = X @ W1^T          [16384,2048]x[2048,256]
  k_gemm_bf16<0><<<dim3(OO / 128, MROWS / 128), 256, 0, stream>>>(
      Xb, W1b, nullptr, Eb, MROWS, OO, DD);

  // 3) inner attention -> ctx
  k_inner_attn<<<NC * NG, 128, 0, stream>>>(Eb, Xb, Ctxb);

  // 4) feats = X + relu(ctx @ trans^T)   [16384,2048]x[2048,2048]
  k_gemm_bf16<1><<<dim3(DD / 128, MROWS / 128), 256, 0, stream>>>(
      Ctxb, Trb, topk, Featsb, MROWS, DD, DD);

  // 5) fa_e = feats @ iw1^T
  k_gemm_bf16<0><<<dim3(OO / 128, MROWS / 128), 256, 0, stream>>>(
      Featsb, IW1b, nullptr, FAEb, MROWS, OO, DD);

  // 6) p_e = proto @ iw2^T    [384(320),2048]x[2048,256]
  k_gemm_bf16<0><<<dim3(OO / 128, PPAD / 128), 256, 0, stream>>>(
      Pb, IW2b, nullptr, PEb, PPAD, OO, DD);

  // 7) inter attention -> out
  k_inter_attn<<<NC, 128, 0, stream>>>(PEb, FAEb, Featsb, out);
}
